// AugmentedLstm_9354438770975
// MI455X (gfx1250) — compile-verified, matrix operand order checked by reading, not run on hardware
//
#include <hip/hip_runtime.h>
#include <hip/hip_bf16.h>
#include <math.h>

// ---------------------------------------------------------------------------
// AugmentedLstm (highway LSTM), B=16 T=2048 D=H=512, weight-stationary
// persistent kernel for MI455X (gfx1250, wave32, WMMA).
//
// Layout: 32 workgroups, each owns 16 output columns (j0..j0+15).
//   LDS per WG: Wt slice for its 6 gate blocks (6*16 cols x 512 K, bf16) = 96KB
//               + 11 staged f32 WMMA tiles + cell state c[16x16] + bias + len.
//   11 waves/WG: wave 0..5  -> pi tile (A = x_t, fp32->bf16)   gates 0..5
//                wave 6..10 -> ps tile (A = h_{t-1}, bf16 buf)  gates 0..4
//   Each wave: 16 x v_wmma_f32_16x16x32_bf16 (K=512) per timestep.
//   Gate math on threads 0..255 (one per (batch m, col n)), then h chunk is
//   written as bf16 to a global buffer; a grid-wide atomic barrier orders
//   steps (h produced at step t is read by all WGs at step t+1).
// ---------------------------------------------------------------------------

typedef __attribute__((ext_vector_type(16))) __bf16 v16bf;
typedef __attribute__((ext_vector_type(8)))  __bf16 v8bf;
typedef __attribute__((ext_vector_type(8)))  float  v8f;

#define B_    16
#define T_    2048
#define D_    512
#define H_    512
#define NWG   32
#define JC    16          // output columns per workgroup
#define NWAVE 11
#define BLOCK (NWAVE * 32)

// LDS carve-up (bytes)
#define WLDS_OFF  0
#define WLDS_SZ   (6 * 16 * D_ * 2)          // 98304: bf16 [gate][n][k]
#define TILE_OFF  (WLDS_OFF + WLDS_SZ)       // 11 tiles of 16x16 f32
#define TILE_SZ   (NWAVE * 256 * 4)          // 11264
#define CST_OFF   (TILE_OFF + TILE_SZ)       // cell state [16x16] f32
#define CST_SZ    (256 * 4)
#define BLDS_OFF  (CST_OFF + CST_SZ)         // bias [6][16] f32
#define BLDS_SZ   (96 * 4)
#define LLDS_OFF  (BLDS_OFF + BLDS_SZ)       // lengths [16] i32
#define LLDS_SZ   (16 * 4)
#define LDS_TOTAL (LLDS_OFF + LLDS_SZ)       // 111040 bytes < 320KB/WGP

__device__ __forceinline__ float sigmoidf_(float v) {
    return 1.0f / (1.0f + __expf(-v));
}

__global__ void lstm_init_state(__bf16* hbuf, unsigned int* bar) {
    int i = blockIdx.x * blockDim.x + threadIdx.x;
    if (i == 0) *bar = 0u;
    for (int k = i; k < B_ * H_; k += blockDim.x * gridDim.x)
        hbuf[k] = (__bf16)0.0f;
}

__global__ void __launch_bounds__(BLOCK)
lstm_persistent(const float* __restrict__ x,      // [B,T,D]
                const int*   __restrict__ lengths,// [B]
                const float* __restrict__ W,      // [6H, D]
                const float* __restrict__ bias,   // [6H]
                float*       __restrict__ out,    // [B,T,H]
                __bf16*      __restrict__ hbuf,   // [B,H] bf16 carry
                unsigned int* __restrict__ bar)   // grid barrier counter
{
    extern __shared__ char smem[];
    __bf16* wlds = (__bf16*)(smem + WLDS_OFF);
    float*  tiles = (float*)(smem + TILE_OFF);
    float*  cst   = (float*)(smem + CST_OFF);
    float*  blds  = (float*)(smem + BLDS_OFF);
    int*    llds  = (int*)  (smem + LLDS_OFF);

    const int tid  = threadIdx.x;
    const int wave = tid >> 5;
    const int lane = tid & 31;
    const int half = lane >> 4;      // lane group 0/1
    const int mrow = lane & 15;      // M index (A) / N index (B)
    const int j0   = blockIdx.x * JC;

    // ---- preload weight slice (fp32 -> bf16) into LDS: wlds[(g*16+n)*512+k]
    for (int idx = tid; idx < 6 * 16 * D_; idx += BLOCK) {
        int r = idx >> 9;            // row within slice
        int k = idx & (D_ - 1);
        int g = r >> 4, n = r & 15;
        wlds[idx] = (__bf16)W[(size_t)(g * H_ + j0 + n) * D_ + k];
    }
    for (int idx = tid; idx < 96; idx += BLOCK) {
        int g = idx >> 4, n = idx & 15;
        blds[idx] = bias[g * H_ + j0 + n];
    }
    if (tid < 16) llds[tid] = lengths[tid];
    for (int idx = tid; idx < 256; idx += BLOCK) cst[idx] = 0.0f;
    __syncthreads();

    // 16-bit A/B operand K-swizzle (ISA 7.12.2): per lane, two contiguous
    // 8-element chunks: [base, base+8) and [base+16, base+24), base = half*8.
    const int klo = half * 8;
    const int khi = half * 8 + 16;

    const bool is_pi = (wave < 6);
    const int  gate  = is_pi ? wave : (wave - 6);
    const __bf16* wbase = wlds + (size_t)(gate * 16 + mrow) * D_;

    unsigned int target = 0;

    for (int t = 0; t < T_; ++t) {
        // ------------------ GEMM phase: one 16x16 tile per wave ------------
        v8f acc = {};
        if (is_pi) {
            const float* xr = x + ((size_t)mrow * T_ + t) * D_;
            __builtin_prefetch(xr + D_, 0, 1);   // next timestep's row
#pragma unroll 4
            for (int kk = 0; kk < D_; kk += 32) {
                const float* p0 = xr + kk + klo;
                const float* p1 = xr + kk + khi;
                v16bf a;
#pragma unroll
                for (int e = 0; e < 8; ++e) {
                    a[e]     = (__bf16)p0[e];
                    a[8 + e] = (__bf16)p1[e];
                }
                v8bf b0 = *(const v8bf*)(wbase + kk + klo);  // ds_load_b128
                v8bf b1 = *(const v8bf*)(wbase + kk + khi);
                v16bf bm = __builtin_shufflevector(b0, b1,
                    0,1,2,3,4,5,6,7,8,9,10,11,12,13,14,15);
                acc = __builtin_amdgcn_wmma_f32_16x16x32_bf16(
                    false, a, false, bm, (short)0, acc, false, false);
            }
        } else {
            const __bf16* hr = hbuf + mrow * H_;
#pragma unroll 4
            for (int kk = 0; kk < D_; kk += 32) {
                v8bf a0 = *(const v8bf*)(hr + kk + klo);     // global b128
                v8bf a1 = *(const v8bf*)(hr + kk + khi);
                v16bf a = __builtin_shufflevector(a0, a1,
                    0,1,2,3,4,5,6,7,8,9,10,11,12,13,14,15);
                v8bf b0 = *(const v8bf*)(wbase + kk + klo);
                v8bf b1 = *(const v8bf*)(wbase + kk + khi);
                v16bf bm = __builtin_shufflevector(b0, b1,
                    0,1,2,3,4,5,6,7,8,9,10,11,12,13,14,15);
                acc = __builtin_amdgcn_wmma_f32_16x16x32_bf16(
                    false, a, false, bm, (short)0, acc, false, false);
            }
        }
        // C/D layout: VGPR r -> M = r + 8*half, N = mrow
        {
            float* tptr = tiles + wave * 256;
#pragma unroll
            for (int r = 0; r < 8; ++r)
                tptr[(r + half * 8) * 16 + mrow] = acc[r];
        }
        __syncthreads();

        // ------------------ gate phase: threads 0..255 ---------------------
        if (tid < 256) {
            const int m = tid >> 4, n = tid & 15;
            const int o = (m << 4) | n;
            float pi5 = tiles[5 * 256 + o] + blds[5 * 16 + n];
            float g0 = tiles[0 * 256 + o] + tiles[6  * 256 + o] + 2.0f * blds[0 * 16 + n];
            float g1 = tiles[1 * 256 + o] + tiles[7  * 256 + o] + 2.0f * blds[1 * 16 + n];
            float g2 = tiles[2 * 256 + o] + tiles[8  * 256 + o] + 2.0f * blds[2 * 16 + n];
            float g3 = tiles[3 * 256 + o] + tiles[9  * 256 + o] + 2.0f * blds[3 * 16 + n];
            float g4 = tiles[4 * 256 + o] + tiles[10 * 256 + o] + 2.0f * blds[4 * 16 + n];

            float ig = sigmoidf_(g0);
            float fg = sigmoidf_(g1);
            float mi = tanhf(g2);
            float og = sigmoidf_(g3);
            float hw = sigmoidf_(g4);

            float c   = cst[o];
            float mem = ig * mi + fg * c;
            float ov  = og * tanhf(mem);
            ov = hw * ov + (1.0f - hw) * pi5;

            float msk = (t < llds[m]) ? 1.0f : 0.0f;
            ov  *= msk;
            mem *= msk;

            cst[o] = mem;                                          // c stays in LDS
            out[((size_t)m * T_ + t) * H_ + (j0 + n)] = ov;        // padded rows -> 0
            hbuf[m * H_ + j0 + n] = (__bf16)ov;                    // carry as bf16
        }
        __threadfence();
        __syncthreads();

        // ------------------ grid-wide step barrier -------------------------
        target += NWG;
        if (tid == 0) {
            __hip_atomic_fetch_add(bar, 1u, __ATOMIC_ACQ_REL,
                                   __HIP_MEMORY_SCOPE_AGENT);
            while (__hip_atomic_load(bar, __ATOMIC_ACQUIRE,
                                     __HIP_MEMORY_SCOPE_AGENT) < target) {
                __builtin_amdgcn_s_sleep(1);
            }
        }
        __syncthreads();
    }
}

extern "C" void kernel_launch(void* const* d_in, const int* in_sizes, int n_in,
                              void* d_out, int out_size, void* d_ws, size_t ws_size,
                              hipStream_t stream) {
    const float* x       = (const float*)d_in[0];   // [B,T,D] f32
    const int*   lengths = (const int*)  d_in[1];   // [B] i32
    const float* W       = (const float*)d_in[2];   // [6H,D] f32
    const float* b       = (const float*)d_in[3];   // [6H]  f32
    float*       out     = (float*)d_out;           // [B,T,H] f32

    __bf16*       hbuf = (__bf16*)d_ws;                                   // 16KB
    unsigned int* bar  = (unsigned int*)((char*)d_ws + B_ * H_ * 2);      // 4B

    hipLaunchKernelGGL(lstm_init_state, dim3(1), dim3(256), 0, stream, hbuf, bar);
    hipLaunchKernelGGL(lstm_persistent, dim3(NWG), dim3(BLOCK), LDS_TOTAL, stream,
                       x, lengths, W, b, out, hbuf, bar);
}
